// YICALlamaDemo_18906446037422
// MI455X (gfx1250) — matrix operand order repaired and checked by reading, not compiled, not run on hardware
//
#include <hip/hip_runtime.h>
#include <hip/hip_bf16.h>

// ---------------- model dims ----------------
#define BATCH   2
#define SEQ     1024
#define HID     1024
#define NLAYERS 6
#define NHEADS  16
#define HDIM    64
#define INTER   2730
#define NVOCAB  32000

// ---------------- WMMA types ----------------
typedef __attribute__((ext_vector_type(16))) __bf16 v16bf;
typedef __attribute__((ext_vector_type(2)))  __bf16 v2bf;
typedef __attribute__((ext_vector_type(2)))  float  v2f;
typedef __attribute__((ext_vector_type(8)))  float  v8f;

union Frag { v16bf v; uint4 q[2]; unsigned u[8]; };

// Packed f32x2 -> bf16x2 conversion: vector form lets clang emit a single
// V_CVT_PK_BF16_F32 (two real sources) instead of cvt + v_mov_b16/v_perm.
__device__ inline unsigned pack_bf16(float a, float b) {
  v2f f = {a, b};
  union { v2bf h; unsigned u; } x;
  x.h = __builtin_convertvector(f, v2bf);
  return x.u;
}

// ---------------------------------------------------------------------------
// Batched GEMM: C[m,n] = sum_k A[m,k] * (TRANSB ? B[n,k] : B[k,n])  (+resid)
// fp32 in/out, bf16 WMMA compute with f32 accumulation.
// Tile 128(M) x 64(N) x 32(K); 256 threads = 8 waves; each wave owns a 32x32
// sub-tile = four independent 16x16 WMMA accumulator chains.
// Double-buffered LDS + register prefetch: ONE barrier per K-chunk; global
// loads for chunk i+1 are issued before the WMMAs of chunk i (latency hiding).
// K-tail (K % 32, only INTER=2730) is peeled out of the main loop.
// Batch z decomposes as (z/innerBatch, z%innerBatch) with separate pointer
// strides, covering per-(batch,head) strided views without copies.
// ---------------------------------------------------------------------------
#define BM 128
#define BN 64
#define BK 32
#define LDSW 20   // dword row stride: 16 bf16-pairs + 4 pad (80B, 16B-aligned)

template<int TRANSB>
__global__ __launch_bounds__(256) void gemm_wmma(
    const float* __restrict__ A, int lda, long sAo, long sAi,
    const float* __restrict__ B, int ldb, long sBo, long sBi,
    float* C, int ldc, long sCo, long sCi,
    const float* resid,
    int M, int N, int K, int innerBatch)
{
  __shared__ __align__(16) unsigned Asl[2][BM][LDSW];
  __shared__ __align__(16) unsigned Bsl[2][BN][LDSW];

  const int z  = blockIdx.z;
  const int zo = z / innerBatch;
  const int zi = z % innerBatch;
  A += zo * sAo + zi * sAi;
  B += zo * sBo + zi * sBi;
  C += zo * sCo + zi * sCi;
  if (resid) resid += zo * sCo + zi * sCi;

  const int n0  = blockIdx.x * BN;
  const int m0  = blockIdx.y * BM;
  const int tid = threadIdx.x;
  const int lane = tid & 31;
  const int wave = tid >> 5;     // 0..7
  const int wm   = wave & 3;     // 32-row group
  const int wn   = wave >> 2;    // 32-col group
  const int hi   = lane >> 4;
  const int llo  = lane & 15;

  const bool fullN = (n0 + BN <= N);
  const bool vecA  = ((lda & 3) == 0);
  const bool vecB  = ((ldb & 3) == 0);

  // staging coordinates: each thread owns 16 contiguous K-floats of one row
  const int ar = tid >> 1, aq = tid & 1;             // A: 128 rows x 2 halves
  const float* aBase = A + (long)(m0 + ar) * lda + 16 * aq;
  const int br = tid >> 1, bq = tid & 1;             // B(T): 64 rows x 2 halves
  const float* bBase = B + (long)(n0 + br) * ldb + 16 * bq;

  float4 fa[4];       // A prefetch regs
  float4 fb[4];       // B prefetch regs (TRANSB)
  float  fbs[8];      // B prefetch regs (gather-transpose path)

  v8f acc00 = {}, acc01 = {}, acc10 = {}, acc11 = {};

  auto loadA = [&](int k0) {
    const float* g = aBase + k0;
    if (vecA) {
      const float4* g4 = (const float4*)g;
      fa[0] = g4[0]; fa[1] = g4[1]; fa[2] = g4[2]; fa[3] = g4[3];
    } else {
      #pragma unroll
      for (int j = 0; j < 4; ++j)
        fa[j] = make_float4(g[4*j], g[4*j+1], g[4*j+2], g[4*j+3]);
    }
  };

  auto loadB = [&](int k0) {
    if (TRANSB) {
      if (tid < 128) {
        if (fullN || (n0 + br) < N) {
          const float* g = bBase + k0;
          if (vecB) {
            const float4* g4 = (const float4*)g;
            fb[0] = g4[0]; fb[1] = g4[1]; fb[2] = g4[2]; fb[3] = g4[3];
          } else {
            #pragma unroll
            for (int j = 0; j < 4; ++j)
              fb[j] = make_float4(g[4*j], g[4*j+1], g[4*j+2], g[4*j+3]);
          }
        } else {
          #pragma unroll
          for (int j = 0; j < 4; ++j) fb[j] = make_float4(0.f, 0.f, 0.f, 0.f);
        }
      }
    } else {
      // gather-transpose (only small P@V GEMM; K%32==0, N==BN there)
      #pragma unroll
      for (int it = 0; it < 4; ++it) {
        int idx = tid + it * 256;
        int r = idx >> 4, pc = idx & 15;
        int k = k0 + 2 * pc, nn = n0 + r;
        bool act = fullN || nn < N;
        fbs[2*it]     = act ? B[(long)k * ldb + nn]       : 0.f;
        fbs[2*it + 1] = act ? B[(long)(k + 1) * ldb + nn] : 0.f;
      }
    }
  };

  auto commit = [&](int buf) {
    uint4 u0, u1;
    u0.x = pack_bf16(fa[0].x, fa[0].y); u0.y = pack_bf16(fa[0].z, fa[0].w);
    u0.z = pack_bf16(fa[1].x, fa[1].y); u0.w = pack_bf16(fa[1].z, fa[1].w);
    u1.x = pack_bf16(fa[2].x, fa[2].y); u1.y = pack_bf16(fa[2].z, fa[2].w);
    u1.z = pack_bf16(fa[3].x, fa[3].y); u1.w = pack_bf16(fa[3].z, fa[3].w);
    *(uint4*)&Asl[buf][ar][8 * aq]     = u0;
    *(uint4*)&Asl[buf][ar][8 * aq + 4] = u1;
    if (TRANSB) {
      if (tid < 128) {
        uint4 v0, v1;
        v0.x = pack_bf16(fb[0].x, fb[0].y); v0.y = pack_bf16(fb[0].z, fb[0].w);
        v0.z = pack_bf16(fb[1].x, fb[1].y); v0.w = pack_bf16(fb[1].z, fb[1].w);
        v1.x = pack_bf16(fb[2].x, fb[2].y); v1.y = pack_bf16(fb[2].z, fb[2].w);
        v1.z = pack_bf16(fb[3].x, fb[3].y); v1.w = pack_bf16(fb[3].z, fb[3].w);
        *(uint4*)&Bsl[buf][br][8 * bq]     = v0;
        *(uint4*)&Bsl[buf][br][8 * bq + 4] = v1;
      }
    } else {
      #pragma unroll
      for (int it = 0; it < 4; ++it) {
        int idx = tid + it * 256;
        int r = idx >> 4, pc = idx & 15;
        Bsl[buf][r][pc] = pack_bf16(fbs[2*it], fbs[2*it + 1]);
      }
    }
  };

  auto mma = [&](int buf) {
    Frag ga0, ga1, gb0, gb1;
    const int mr0 = wm * 32 + llo, mr1 = mr0 + 16;
    ga0.q[0] = *(const uint4*)&Asl[buf][mr0][8 * hi];
    ga0.q[1] = *(const uint4*)&Asl[buf][mr0][8 * hi + 4];
    ga1.q[0] = *(const uint4*)&Asl[buf][mr1][8 * hi];
    ga1.q[1] = *(const uint4*)&Asl[buf][mr1][8 * hi + 4];
    const int nr0 = wn * 32 + llo, nr1 = nr0 + 16;
    gb0.q[0] = *(const uint4*)&Bsl[buf][nr0][8 * hi];
    gb0.q[1] = *(const uint4*)&Bsl[buf][nr0][8 * hi + 4];
    gb1.q[0] = *(const uint4*)&Bsl[buf][nr1][8 * hi];
    gb1.q[1] = *(const uint4*)&Bsl[buf][nr1][8 * hi + 4];

    acc00 = __builtin_amdgcn_wmma_f32_16x16x32_bf16(
        false, ga0.v, false, gb0.v, (short)0, acc00, false, false);
    acc01 = __builtin_amdgcn_wmma_f32_16x16x32_bf16(
        false, ga0.v, false, gb1.v, (short)0, acc01, false, false);
    acc10 = __builtin_amdgcn_wmma_f32_16x16x32_bf16(
        false, ga1.v, false, gb0.v, (short)0, acc10, false, false);
    acc11 = __builtin_amdgcn_wmma_f32_16x16x32_bf16(
        false, ga1.v, false, gb1.v, (short)0, acc11, false, false);
  };

  // -------- main loop over full K-chunks: double buffer, 1 barrier/iter ----
  const int kFull   = K & ~(BK - 1);
  const int nChunks = kFull / BK;

  if (nChunks > 0) {
    loadA(0); loadB(0);
    commit(0);
    for (int i = 0; i < nChunks; ++i) {
      __syncthreads();                       // buffer (i&1) ready for reads
      const bool hasNext = (i + 1 < nChunks);
      if (hasNext) { loadA((i + 1) * BK); loadB((i + 1) * BK); }  // prefetch
      mma(i & 1);                            // math on current buffer
      if (hasNext) commit((i + 1) & 1);      // convert+store into other buffer
    }
  }

  // -------- peeled K-tail (only K = 2730 path), guarded scalar staging -----
  if (kFull < K) {
    const int buf = nChunks & 1;
    __syncthreads();
    #pragma unroll
    for (int it = 0; it < 8; ++it) {
      int idx = tid + it * 256;
      int r = idx >> 4, pc = idx & 15;
      int k = kFull + 2 * pc;
      const float* arow = A + (long)(m0 + r) * lda;
      float a0 = (k     < K) ? arow[k]     : 0.f;
      float a1 = (k + 1 < K) ? arow[k + 1] : 0.f;
      Asl[buf][r][pc] = pack_bf16(a0, a1);
    }
    #pragma unroll
    for (int it = 0; it < 4; ++it) {
      int idx = tid + it * 256;
      int r = idx >> 4, pc = idx & 15;
      int k = kFull + 2 * pc, nn = n0 + r;
      float b0 = 0.f, b1 = 0.f;
      if (fullN || nn < N) {
        if (TRANSB) {
          if (k     < K) b0 = B[(long)nn * ldb + k];
          if (k + 1 < K) b1 = B[(long)nn * ldb + k + 1];
        } else {
          if (k     < K) b0 = B[(long)k * ldb + nn];
          if (k + 1 < K) b1 = B[(long)(k + 1) * ldb + nn];
        }
      }
      Bsl[buf][r][pc] = pack_bf16(b0, b1);
    }
    __syncthreads();
    mma(buf);
  }

  // -------- store (C layout: elem r, lane l -> row r+8*hi, col l&15) -------
  const int col0 = n0 + wn * 32 + llo;
  const int col1 = col0 + 16;
  const int rbase0 = m0 + wm * 32;
  #pragma unroll
  for (int r = 0; r < 8; ++r) {
    int row0 = rbase0 + r + 8 * hi;
    int row1 = row0 + 16;
    if (fullN || col0 < N) {
      float v0 = acc00[r], v1 = acc10[r];
      if (resid) { v0 += resid[(long)row0 * ldc + col0]; v1 += resid[(long)row1 * ldc + col0]; }
      C[(long)row0 * ldc + col0] = v0;
      C[(long)row1 * ldc + col0] = v1;
    }
    if (fullN || col1 < N) {
      float v0 = acc01[r], v1 = acc11[r];
      if (resid) { v0 += resid[(long)row0 * ldc + col1]; v1 += resid[(long)row1 * ldc + col1]; }
      C[(long)row0 * ldc + col1] = v0;
      C[(long)row1 * ldc + col1] = v1;
    }
  }
}

// ---------------------------------------------------------------------------
// h[row,:] = tok_emb[ids[row],:] + pos_emb[row % SEQ,:]
// ---------------------------------------------------------------------------
__global__ void embed_kernel(const int* __restrict__ ids,
                             const float* __restrict__ tok_emb,
                             const float* __restrict__ pos_emb,
                             float* __restrict__ h)
{
  int row = blockIdx.x;
  long tok = ids[row];
  int  pos = row % SEQ;
  const float* te = tok_emb + tok * HID;
  const float* pe = pos_emb + (long)pos * HID;
  float* hr = h + (long)row * HID;
  for (int c = threadIdx.x; c < HID; c += blockDim.x)
    hr[c] = te[c] + pe[c];
}

// ---------------------------------------------------------------------------
// LayerNorm over last dim (HID), one block per row
// ---------------------------------------------------------------------------
__global__ void layernorm_kernel(const float* __restrict__ x,
                                 const float* __restrict__ w,
                                 const float* __restrict__ b,
                                 float* __restrict__ out)
{
  __shared__ float red[256];
  int row = blockIdx.x;
  const float* xr = x + (long)row * HID;
  float* outr = out + (long)row * HID;

  float s = 0.f;
  for (int i = threadIdx.x; i < HID; i += 256) s += xr[i];
  red[threadIdx.x] = s; __syncthreads();
  for (int o = 128; o > 0; o >>= 1) {
    if (threadIdx.x < o) red[threadIdx.x] += red[threadIdx.x + o];
    __syncthreads();
  }
  float mu = red[0] * (1.0f / HID);
  __syncthreads();

  float v = 0.f;
  for (int i = threadIdx.x; i < HID; i += 256) {
    float d = xr[i] - mu; v += d * d;
  }
  red[threadIdx.x] = v; __syncthreads();
  for (int o = 128; o > 0; o >>= 1) {
    if (threadIdx.x < o) red[threadIdx.x] += red[threadIdx.x + o];
    __syncthreads();
  }
  float rstd = rsqrtf(red[0] * (1.0f / HID) + 1e-5f);

  for (int i = threadIdx.x; i < HID; i += 256)
    outr[i] = (xr[i] - mu) * rstd * w[i] + b[i];
}

// ---------------------------------------------------------------------------
// softmax over row of length SEQ with pre-scale; grid(SEQ, BATCH*NHEADS)
// ---------------------------------------------------------------------------
__global__ void softmax_kernel(float* __restrict__ scores, float scale)
{
  __shared__ float red[256];
  float* r = scores + ((long)blockIdx.y * SEQ + blockIdx.x) * SEQ;

  float mx = -3.0e38f;
  for (int i = threadIdx.x; i < SEQ; i += 256)
    mx = fmaxf(mx, r[i] * scale);
  red[threadIdx.x] = mx; __syncthreads();
  for (int o = 128; o > 0; o >>= 1) {
    if (threadIdx.x < o) red[threadIdx.x] = fmaxf(red[threadIdx.x], red[threadIdx.x + o]);
    __syncthreads();
  }
  mx = red[0];
  __syncthreads();

  float s = 0.f;
  for (int i = threadIdx.x; i < SEQ; i += 256) {
    float e = __expf(r[i] * scale - mx);
    r[i] = e;
    s += e;
  }
  red[threadIdx.x] = s; __syncthreads();
  for (int o = 128; o > 0; o >>= 1) {
    if (threadIdx.x < o) red[threadIdx.x] += red[threadIdx.x + o];
    __syncthreads();
  }
  float inv = 1.0f / red[0];
  for (int i = threadIdx.x; i < SEQ; i += 256) r[i] *= inv;
}

// ---------------------------------------------------------------------------
// gate = silu(gate) * up, elementwise
// ---------------------------------------------------------------------------
__global__ void silu_mul_kernel(float* __restrict__ gate,
                                const float* __restrict__ up, long n)
{
  long i = (long)blockIdx.x * blockDim.x + threadIdx.x;
  if (i < n) {
    float g = gate[i];
    gate[i] = (g / (1.0f + __expf(-g))) * up[i];
  }
}

// ---------------------------------------------------------------------------
extern "C" void kernel_launch(void* const* d_in, const int* in_sizes, int n_in,
                              void* d_out, int out_size, void* d_ws, size_t ws_size,
                              hipStream_t stream)
{
  (void)in_sizes; (void)n_in; (void)out_size; (void)ws_size;

  const int*   ids     = (const int*)  d_in[0];
  const float* tok_emb = (const float*)d_in[1];
  const float* pos_emb = (const float*)d_in[2];
  const float* qkv_w   = (const float*)d_in[3];
  const float* o_w     = (const float*)d_in[4];
  const float* gate_w  = (const float*)d_in[5];
  const float* up_w    = (const float*)d_in[6];
  const float* down_w  = (const float*)d_in[7];
  const float* ln1_w   = (const float*)d_in[8];
  const float* ln1_b   = (const float*)d_in[9];
  const float* ln2_w   = (const float*)d_in[10];
  const float* ln2_b   = (const float*)d_in[11];
  const float* lnf_w   = (const float*)d_in[12];
  const float* lnf_b   = (const float*)d_in[13];
  const float* head_w  = (const float*)d_in[14];
  float* out = (float*)d_out;

  const int M = BATCH * SEQ;                 // 2048 token rows
  const float scale = 0.125f;                // HDIM^-0.5 = 1/8

  // workspace layout (floats)
  float* ws   = (float*)d_ws;
  float* h    = ws;  ws += (long)M * HID;          //  8 MB
  float* x    = ws;  ws += (long)M * HID;          //  8 MB
  float* qkv  = ws;  ws += (long)M * 3 * HID;      // 24 MB
  float* ctx  = ws;  ws += (long)M * HID;          //  8 MB
  float* gate = ws;  ws += (long)M * INTER;        // ~21 MB
  float* up   = ws;  ws += (long)M * INTER;        // ~21 MB
  float* sc   = ws;                                 // B*H*S*S = 128 MB

  dim3 blk(256);

  embed_kernel<<<M, blk, 0, stream>>>(ids, tok_emb, pos_emb, h);

  for (int l = 0; l < NLAYERS; ++l) {
    const float* qkvW = qkv_w  + (long)l * 3 * HID * HID;
    const float* oW   = o_w    + (long)l * HID * HID;
    const float* gW   = gate_w + (long)l * INTER * HID;
    const float* uW   = up_w   + (long)l * INTER * HID;
    const float* dW   = down_w + (long)l * HID * INTER;
    const float* w1 = ln1_w + (long)l * HID; const float* b1 = ln1_b + (long)l * HID;
    const float* w2 = ln2_w + (long)l * HID; const float* b2 = ln2_b + (long)l * HID;

    // x = LN1(h)
    layernorm_kernel<<<M, blk, 0, stream>>>(h, w1, b1, x);

    // qkv = x @ qkvW^T   (M x 3HID)
    gemm_wmma<1><<<dim3(3 * HID / BN, M / BM, 1), blk, 0, stream>>>(
        x, HID, 0, 0,  qkvW, HID, 0, 0,  qkv, 3 * HID, 0, 0,
        nullptr, M, 3 * HID, HID, 1);

    // scores[b,hd] = Q @ K^T   (batched over B*NHEADS via strided views)
    gemm_wmma<1><<<dim3(SEQ / BN, SEQ / BM, BATCH * NHEADS), blk, 0, stream>>>(
        qkv,       3 * HID, (long)SEQ * 3 * HID, HDIM,                // Q view
        qkv + HID, 3 * HID, (long)SEQ * 3 * HID, HDIM,                // K view
        sc, SEQ, (long)NHEADS * SEQ * SEQ, (long)SEQ * SEQ,
        nullptr, SEQ, SEQ, HDIM, NHEADS);

    softmax_kernel<<<dim3(SEQ, BATCH * NHEADS), blk, 0, stream>>>(sc, scale);

    // ctx[b,hd] = P @ V   (B non-transposed: V is K x N view)
    gemm_wmma<0><<<dim3(1, SEQ / BM, BATCH * NHEADS), blk, 0, stream>>>(
        sc,            SEQ,     (long)NHEADS * SEQ * SEQ, (long)SEQ * SEQ,
        qkv + 2 * HID, 3 * HID, (long)SEQ * 3 * HID,      HDIM,       // V view
        ctx,           HID,     (long)SEQ * HID,          HDIM,
        nullptr, SEQ, HDIM, SEQ, NHEADS);

    // h = h + ctx @ oW^T
    gemm_wmma<1><<<dim3(HID / BN, M / BM, 1), blk, 0, stream>>>(
        ctx, HID, 0, 0,  oW, HID, 0, 0,  h, HID, 0, 0,
        h, M, HID, HID, 1);

    // x = LN2(h)
    layernorm_kernel<<<M, blk, 0, stream>>>(h, w2, b2, x);

    // gate = x @ gW^T ; up = x @ uW^T   (N = 2730 -> edge tiles guarded)
    gemm_wmma<1><<<dim3((INTER + BN - 1) / BN, M / BM, 1), blk, 0, stream>>>(
        x, HID, 0, 0,  gW, HID, 0, 0,  gate, INTER, 0, 0,
        nullptr, M, INTER, HID, 1);
    gemm_wmma<1><<<dim3((INTER + BN - 1) / BN, M / BM, 1), blk, 0, stream>>>(
        x, HID, 0, 0,  uW, HID, 0, 0,  up, INTER, 0, 0,
        nullptr, M, INTER, HID, 1);

    long nf = (long)M * INTER;
    silu_mul_kernel<<<(unsigned)((nf + 255) / 256), blk, 0, stream>>>(gate, up, nf);

    // h = h + (silu(gate)*up) @ dW^T   (K = 2730 -> peeled guarded tail)
    gemm_wmma<1><<<dim3(HID / BN, M / BM, 1), blk, 0, stream>>>(
        gate, INTER, 0, 0,  dW, INTER, 0, 0,  h, HID, 0, 0,
        h, M, HID, INTER, 1);
  }

  // x = LNf(h); logits = x @ head_w^T
  layernorm_kernel<<<M, blk, 0, stream>>>(h, lnf_w, lnf_b, x);
  gemm_wmma<1><<<dim3(NVOCAB / BN, M / BM, 1), blk, 0, stream>>>(
      x, HID, 0, 0,  head_w, HID, 0, 0,  out, NVOCAB, 0, 0,
      nullptr, M, NVOCAB, HID, 1);
}
